// QuIPA2Linear_25744033972596
// MI455X (gfx1250) — matrix operand order repaired and checked by reading, not compiled, not run on hardware
//
#include <hip/hip_runtime.h>
#include <stdint.h>

typedef __attribute__((ext_vector_type(16))) _Float16 v16h;
typedef __attribute__((ext_vector_type(8)))  float    v8f;
typedef __attribute__((ext_vector_type(2)))  unsigned long long v2u64;

#define M_BATCH 256
#define N_OUT   4096
#define K_IN    4096
#define NGRP    (K_IN / 4)
#define BK      64
#define NSTEP   (K_IN / BK)
#define BN      32

// Raw D4 half-codebook, stored as 2x the actual values (all integers in [-4,4]).
constexpr signed char kD4Raw[512] = {
     0, 0, 0, 0,  -4, 0, 0, 0,  -2,-2,-2,-2,  -2,-2,-2, 0,  -2,-2,-2, 2,  -2,-2, 0,-2,  -2,-2, 0, 0,  -2,-2, 0, 2,
    -2,-2, 2,-2,  -2,-2, 2, 0,  -2,-2, 2, 2,  -2, 0,-2,-2,  -2, 0,-2, 0,  -2, 0,-2, 2,  -2, 0, 0,-2,  -2, 0, 0, 0,
    -2, 0, 0, 2,  -2, 0, 2,-2,  -2, 0, 2, 0,  -2, 0, 2, 2,  -2, 2,-2,-2,  -2, 2,-2, 0,  -2, 2,-2, 2,  -2, 2, 0,-2,
    -2, 2, 0, 0,  -2, 2, 0, 2,  -2, 2, 2,-2,  -2, 2, 2, 0,  -2, 2, 2, 2,   0,-4, 0, 0,   0,-2,-2,-2,   0,-2,-2, 0,
     0,-2,-2, 2,   0,-2, 0,-2,   0,-2, 0, 0,   0,-2, 0, 2,   0,-2, 2,-2,   0,-2, 2, 0,   0,-2, 2, 2,   0, 0,-4, 0,
     0, 0,-2,-2,   0, 0,-2, 0,   0, 0,-2, 2,   0, 0, 0,-4,   0, 0, 0,-2,
    -3,-1,-3,-1,  -3,-1,-3, 1,  -3,-1,-1,-3,  -3,-1,-1,-1,  -3,-1,-1, 1,  -3,-1,-1, 3,  -3,-1, 1,-3,  -3,-1, 1,-1,
    -3,-1, 1, 1,  -3,-1, 1, 3,  -3,-1, 3,-1,  -3,-1, 3, 1,  -3, 1,-3,-1,  -3, 1,-3, 1,  -3, 1,-1,-3,  -3, 1,-1,-1,
    -3, 1,-1, 1,  -3, 1,-1, 3,  -3, 1, 1,-3,  -3, 1, 1,-1,  -3, 1, 1, 1,  -3, 1, 1, 3,  -3, 1, 3,-1,  -3, 1, 3, 1,
    -3, 3,-1,-1,  -3, 3, 1,-1,  -3, 3, 1, 1,  -1,-3,-3,-1,  -1,-3,-3, 1,  -1,-3,-1,-3,  -1,-3,-1,-1,  -1,-3,-1, 1,
    -1,-3,-1, 3,  -1,-3, 1,-3,  -1,-3, 1,-1,  -1,-3, 1, 1,  -1,-3, 1, 3,  -1,-3, 3,-1,  -1,-3, 3, 1,  -1,-1,-3,-3,
    -1,-1,-3,-1,  -1,-1,-3, 1,  -1,-1,-3, 3,  -1,-1,-1,-3,  -1,-1,-1,-1,  -1,-1,-1, 1,  -1,-1,-1, 3,  -1,-1, 1,-3,
    -1,-1, 1,-1,  -1,-1, 1, 1,  -1,-1, 1, 3,  -1,-1, 3,-3,  -1,-1, 3,-1,  -1,-1, 3, 1,  -1,-1, 3, 3,  -1, 1,-3,-3,
    -1, 1,-3,-1,  -1, 1,-3, 1,  -1, 1,-3, 3,  -1, 1,-1,-3,  -1, 1,-1,-1,  -1, 1,-1, 1,  -1, 1,-1, 3,  -1, 1, 1,-3,
    -1, 1, 1,-1,  -1, 1, 1, 1,  -1, 1, 1, 3,  -1, 1, 3,-3,  -1, 1, 3,-1,  -1, 1, 3, 1,  -1, 1, 3, 3,  -1, 3,-3,-1,
    -1, 3,-3, 1,  -1, 3,-1,-3,  -1, 3,-1,-1,  -1, 3,-1, 1,  -1, 3,-1, 3,  -1, 3, 1,-3,  -1, 3, 1,-1,  -1, 3, 1, 1,
    -1, 3, 1, 3,  -1, 3, 3,-1,  -1, 3, 3, 1
};

// f16 bit pattern of raw/2: half-integers are exact. |raw|: 0,1,2,3,4 -> 0,0.5,1,1.5,2
constexpr unsigned short f16_of_raw(int r) {
    int a = r < 0 ? -r : r;
    unsigned short m = (a == 0) ? 0x0000
                     : (a == 1) ? 0x3800
                     : (a == 2) ? 0x3C00
                     : (a == 3) ? 0x3E00
                     :            0x4000;
    return (unsigned short)(r < 0 ? (m | 0x8000) : m);
}

struct Codebook { unsigned long long v[128]; };
constexpr Codebook make_codebook() {
    Codebook c{};
    for (int i = 0; i < 128; ++i) {
        unsigned long long p = 0;
        for (int j = 0; j < 4; ++j)
            p |= (unsigned long long)f16_of_raw(kD4Raw[i * 4 + j]) << (16 * j);
        c.v[i] = p;
    }
    return c;
}
__device__ __constant__ Codebook g_cb = make_codebook();

// Pre-pass: fold diagonal + group scales into activations, convert to f16.
__global__ __launch_bounds__(256) void scale_convert_x(const float* __restrict__ x,
                                                       const float* __restrict__ sw,
                                                       const float* __restrict__ qs,
                                                       _Float16* __restrict__ xh) {
    int idx = blockIdx.x * blockDim.x + threadIdx.x;
    int k = idx & (K_IN - 1);
    xh[idx] = (_Float16)(x[idx] * sw[k] * qs[k >> 2]);
}

__device__ __forceinline__ unsigned long long dq(const unsigned long long* cb, int idx) {
    unsigned long long w = cb[idx & 127];
    if (idx & 128) w ^= 0x8000800080008000ull;   // negate 4 packed f16
    return w;
}

// One workgroup = 8 waves, covers all 256 batch rows x 32 output columns.
// Wave w owns M rows [32w, 32w+32) and all 32 N columns as a 2x2 grid of
// 16x16 WMMA tiles -> 8 v_wmma_f32_16x16x32_f16 per 64-wide K step, with each
// A/B fragment feeding two WMMAs. The f16 weight tile (32n x 64k) is decoded
// into a double-buffered LDS tile: one barrier per step, next stage's decode
// overlaps the current WMMAs.
__global__ __launch_bounds__(256) void quip_wmma_gemm(const _Float16* __restrict__ Xh,
                                                      const int* __restrict__ Qidxs,
                                                      float* __restrict__ Out) {
    __shared__ unsigned long long s_cb[128];     // packed f16x4 codebook (1 KB)
    __shared__ _Float16 s_B[2][BN * BK];         // column-major [n][k], ping-pong (2x4 KB)

    const int tid  = threadIdx.x;
    const int wave = tid >> 5;
    const int lane = tid & 31;
    const int h    = lane >> 4;      // lane half (0/1)
    const int ln   = lane & 15;
    const int n0   = blockIdx.x * BN;

    if (tid < 128) s_cb[tid] = g_cb.v[tid];

    // Staging: thread -> (column sn, adjacent group pair sg..sg+1) of the 64-wide step
    const int sn = tid >> 3;         // 0..31
    const int sg = (tid & 7) * 2;    // 0,2,..,14
    const int* qp = Qidxs + (size_t)(n0 + sn) * NGRP + sg;
    _Float16* sbst = &s_B[0][sn * BK + sg * 4];  // 16B-aligned staging slot (buf 0)

    v8f acc00 = {}, acc01 = {}, acc10 = {}, acc11 = {};
    const int m0 = wave * 32;        // this wave's first batch row

    // Prologue: stage step 0 into buffer 0.
    int2 idx = *(const int2*)qp;
    __syncthreads();                 // s_cb ready
    {
        v2u64 p; p.x = dq(s_cb, idx.x); p.y = dq(s_cb, idx.y);
        *(v2u64*)sbst = p;
    }
    int2 idxn = *(const int2*)(qp + 16);
    __syncthreads();                 // buffer 0 ready

    for (int ks = 0; ks < NSTEP; ++ks) {
        const int cur = ks & 1;
        const int nxt = cur ^ 1;

        // ---- decode next step's 32x64 weight tile into the other buffer ----
        if (ks + 1 < NSTEP) {
            v2u64 p; p.x = dq(s_cb, idxn.x); p.y = dq(s_cb, idxn.y);
            *(v2u64*)(sbst + nxt * (BN * BK)) = p;
            if (ks + 2 < NSTEP) idxn = *(const int2*)(qp + (size_t)(ks + 2) * 16);
        }

        // ---- compute current step ----
        const int k0 = ks * BK;
        const _Float16* sb = &s_B[cur][0];
        #pragma unroll
        for (int kk = 0; kk < BK; kk += 32) {
            // B fragments: lane (h, ln=n): element e -> K = 16h + e (contiguous in LDS)
            v16h b0 = *(const v16h*)(sb + ln * BK + kk + 16 * h);
            v16h b1 = *(const v16h*)(sb + (16 + ln) * BK + kk + 16 * h);

            // A fragments: lane (h, ln=m): VGPR v -> K=16h+2v(+c)
            const _Float16* xr0 = Xh + (size_t)(m0 + ln) * K_IN + k0 + kk + 16 * h;
            const _Float16* xr1 = xr0 + (size_t)16 * K_IN;
            union { v16h v; unsigned u[8]; } a0, a1;
            #pragma unroll
            for (int v = 0; v < 4; ++v) {
                a0.u[v]     = *(const unsigned*)(xr0 + 2 * v);
                a0.u[4 + v] = *(const unsigned*)(xr0 + 8 + 2 * v);
                a1.u[v]     = *(const unsigned*)(xr1 + 2 * v);
                a1.u[4 + v] = *(const unsigned*)(xr1 + 8 + 2 * v);
            }

            acc00 = __builtin_amdgcn_wmma_f32_16x16x32_f16(false, a0.v, false, b0,
                                                           (short)0, acc00, false, false);
            acc01 = __builtin_amdgcn_wmma_f32_16x16x32_f16(false, a0.v, false, b1,
                                                           (short)0, acc01, false, false);
            acc10 = __builtin_amdgcn_wmma_f32_16x16x32_f16(false, a1.v, false, b0,
                                                           (short)0, acc10, false, false);
            acc11 = __builtin_amdgcn_wmma_f32_16x16x32_f16(false, a1.v, false, b1,
                                                           (short)0, acc11, false, false);
        }
        __syncthreads();   // nxt-buffer stores visible; cur-buffer reads done
    }

    // D layout: lane (h, ln=n), VGPR r -> M = r + 8h
    float* ob = Out + (size_t)(m0 + 8 * h) * N_OUT + n0 + ln;
    #pragma unroll
    for (int r = 0; r < 8; ++r) {
        ob[(size_t)r * N_OUT]            = acc00[r];
        ob[(size_t)r * N_OUT + 16]       = acc01[r];
        ob[(size_t)(16 + r) * N_OUT]     = acc10[r];
        ob[(size_t)(16 + r) * N_OUT + 16] = acc11[r];
    }
}

extern "C" void kernel_launch(void* const* d_in, const int* in_sizes, int n_in,
                              void* d_out, int out_size, void* d_ws, size_t ws_size,
                              hipStream_t stream) {
    (void)in_sizes; (void)n_in; (void)out_size; (void)ws_size;
    const float* input = (const float*)d_in[0];
    const float* sw    = (const float*)d_in[1];
    const float* qs    = (const float*)d_in[2];
    const int*   qidx  = (const int*)d_in[3];
    float*    out = (float*)d_out;
    _Float16* xh  = (_Float16*)d_ws;   // 256*4096 f16 = 2 MB scratch

    scale_convert_x<<<dim3((M_BATCH * K_IN) / 256), dim3(256), 0, stream>>>(input, sw, qs, xh);
    quip_wmma_gemm<<<dim3(N_OUT / BN), dim3(256), 0, stream>>>(xh, qidx, out);
}
